// MAPPOPruningController_73624329388250
// MI455X (gfx1250) — compile-verified, matrix operand order checked by reading, not run on hardware
//
#include <hip/hip_runtime.h>
#include <hip/hip_bf16.h>

typedef _Float16 half_t;
typedef __attribute__((ext_vector_type(16))) _Float16 v16h;
typedef __attribute__((ext_vector_type(8)))  _Float16 v8h;
typedef __attribute__((ext_vector_type(8)))  float    v8f;
typedef __attribute__((ext_vector_type(4)))  int      i32x4;

#define EMBED 768
#define HIDDEN 384
#define NHEAD 12
#define HDIM 64
#define BATCH 8
#define NTOK 1024
#define ROWS (BATCH * NTOK)          // 8192
#define MIN_KEEP 102                 // max(int(1024*0.1),1)
#define QKVC (3 * EMBED)             // 2304

#if defined(__has_builtin)
#if __has_builtin(__builtin_amdgcn_global_load_async_to_lds_b128) && \
    __has_builtin(__builtin_amdgcn_s_wait_asynccnt)
#define HAVE_ASYNC_LDS 1
#endif
#endif

// 16-byte global -> LDS copy. CDNA5 path: GLOBAL_LOAD_ASYNC_TO_LDS_B128
// (per-lane global & LDS addresses, tracked by ASYNCcnt). The builtin takes
// generic i32x4 pointers (clang inserts the AS1/AS3 addrspacecasts itself).
// Fallback: synchronous copy.
__device__ __forceinline__ void copy16_to_lds(const half_t* __restrict__ g,
                                              half_t* l) {
#if defined(HAVE_ASYNC_LDS)
  __builtin_amdgcn_global_load_async_to_lds_b128((i32x4*)g, (i32x4*)l, 0, 0);
#else
  *(i32x4*)l = *(const i32x4*)g;
#endif
}

__device__ __forceinline__ void wait_lds_copies() {
#if defined(HAVE_ASYNC_LDS)
  __builtin_amdgcn_s_wait_asynccnt(0);
#endif
  __syncthreads();
}

// ---------------------------------------------------------------------------
// WMMA fragment helper. 16-bit A-matrix 16x32 layout (ISA 7.12.2):
// lane L holds row M = L%16; element e of the v16h maps to
//   K = (e<8 ? 0 : 16) + 8*(L/16) + (e&7)
// B (32x16) uses the same element->K mapping with lane L holding column L%16.
// ---------------------------------------------------------------------------
__device__ __forceinline__ v16h frag_from_f16(const half_t* __restrict__ p) {
  v8h lo = *(const v8h*)p;
  v8h hi = *(const v8h*)(p + 16);
  v16h r;
#pragma unroll
  for (int i = 0; i < 8; ++i) { r[i] = lo[i]; r[i + 8] = hi[i]; }
  return r;
}

// ---------------------------------------------------------------------------
// 1) per-batch mean over tokens
// ---------------------------------------------------------------------------
__global__ __launch_bounds__(256) void rowmean_kernel(const float* __restrict__ tokens,
                                                      float* __restrict__ tmean) {
  int b = blockIdx.x;
  for (int c = threadIdx.x; c < EMBED; c += 256) {
    float s = 0.f;
    for (int n = 0; n < NTOK; ++n)
      s += tokens[((size_t)(b * NTOK + n)) * EMBED + c];
    tmean[b * EMBED + c] = s * (1.f / NTOK);
  }
}

// ---------------------------------------------------------------------------
// 2) ctx = tokens + 2*emb + tmean ; LayerNorm ; store f16
// ---------------------------------------------------------------------------
__global__ __launch_bounds__(256) void ln_kernel(const float* __restrict__ tokens,
                                                 const float* __restrict__ tmean,
                                                 const float* __restrict__ layer_emb,
                                                 const int* __restrict__ lidx,
                                                 const float* __restrict__ g,
                                                 const float* __restrict__ bta,
                                                 half_t* __restrict__ xln) {
  __shared__ float red[256];
  int row = blockIdx.x;
  int b = row >> 10;
  int t = threadIdx.x;
  const float* emb = layer_emb + (size_t)lidx[0] * EMBED;

  int c0 = t, c1 = t + 256, c2 = t + 512;
  const float* tr = tokens + (size_t)row * EMBED;
  const float* tm = tmean + b * EMBED;
  float x0 = tr[c0] + 2.f * emb[c0] + tm[c0];
  float x1 = tr[c1] + 2.f * emb[c1] + tm[c1];
  float x2 = tr[c2] + 2.f * emb[c2] + tm[c2];

  red[t] = x0 + x1 + x2;
  __syncthreads();
  for (int s = 128; s > 0; s >>= 1) { if (t < s) red[t] += red[t + s]; __syncthreads(); }
  float mu = red[0] * (1.f / EMBED);
  __syncthreads();
  float d0 = x0 - mu, d1 = x1 - mu, d2 = x2 - mu;
  red[t] = d0 * d0 + d1 * d1 + d2 * d2;
  __syncthreads();
  for (int s = 128; s > 0; s >>= 1) { if (t < s) red[t] += red[t + s]; __syncthreads(); }
  float rs = rsqrtf(red[0] * (1.f / EMBED) + 1e-5f);

  half_t* xr = xln + (size_t)row * EMBED;
  xr[c0] = (half_t)(d0 * rs * g[c0] + bta[c0]);
  xr[c1] = (half_t)(d1 * rs * g[c1] + bta[c1]);
  xr[c2] = (half_t)(d2 * rs * g[c2] + bta[c2]);
}

// ---------------------------------------------------------------------------
// elementwise f32 -> f16 cast (tokens and weights)
// ---------------------------------------------------------------------------
__global__ __launch_bounds__(256) void cast_kernel(const float* __restrict__ in,
                                                   half_t* __restrict__ out, int n) {
  int i = blockIdx.x * 256 + threadIdx.x;
  if (i < n) out[i] = (half_t)in[i];
}

// ---------------------------------------------------------------------------
// WMMA GEMM: out[M,Nout] = A[M,K](f16) @ Wh[Nout,K].T(f16) + bias
// 256 threads = 8 waves; block computes 128 rows x 16 cols.
// The 16-col weight strip (contiguous 16*K f16) is staged in LDS once per
// block via async global->LDS b128 copies; B fragments then come from LDS.
// ---------------------------------------------------------------------------
template <bool GELU, bool OUTF16, bool MASKQ, int K>
__global__ __launch_bounds__(256) void gemm_kernel(const half_t* __restrict__ A,
                                                   const half_t* __restrict__ Wh,
                                                   const float* __restrict__ bias,
                                                   half_t* __restrict__ outh,
                                                   float* __restrict__ outf,
                                                   const float* __restrict__ maskf,
                                                   int M, int Nout) {
  __shared__ __align__(16) half_t WT[16 * K];
  int tid = threadIdx.x;
  int wave = tid >> 5, lane = tid & 31;
  int lm = lane & 15, hs = lane >> 4;
  int m0 = blockIdx.y * 128 + wave * 16;
  int n0 = blockIdx.x * 16;

  // stage weight strip: rows n0..n0+15 of Wh are one contiguous 16*K block
  {
    const half_t* wsrc = Wh + (size_t)n0 * K;
    const int nvec = (16 * K) / 8;            // 16B chunks
    for (int i = tid; i < nvec; i += 256)
      copy16_to_lds(wsrc + i * 8, WT + i * 8);
    wait_lds_copies();
  }

  const half_t* arow = A + (size_t)(m0 + lm) * K + 8 * hs;
  const half_t* wl = WT + lm * K + 8 * hs;

  v8f acc = {};
  for (int k0 = 0; k0 < K; k0 += 32) {
    v16h a = frag_from_f16(arow + k0);
    v16h b = frag_from_f16(wl + k0);
    acc = __builtin_amdgcn_wmma_f32_16x16x32_f16(false, a, false, b, (short)0, acc,
                                                 false, false);
  }

  float bv = bias[n0 + lm];
#pragma unroll
  for (int r = 0; r < 8; ++r) {
    int m = m0 + r + 8 * hs;
    float v = acc[r] + bv;
    if (GELU) v = 0.5f * v * (1.0f + erff(v * 0.70710678118654752f));
    if (MASKQ) v *= maskf[m];
    size_t o = (size_t)m * Nout + n0 + lm;
    if (OUTF16) outh[o] = (half_t)v; else outf[o] = v;
  }
}

// ---------------------------------------------------------------------------
// logits = H2 @ w3.T + b3 ; probs = sigmoid
// ---------------------------------------------------------------------------
__global__ __launch_bounds__(128) void logits_kernel(const half_t* __restrict__ H2,
                                                     const float* __restrict__ w3,
                                                     const float* __restrict__ b3,
                                                     float* __restrict__ probs) {
  __shared__ float red[128];
  int row = blockIdx.x, t = threadIdx.x;
  const half_t* hr = H2 + (size_t)row * HIDDEN;
  float s = (float)hr[t] * w3[t] + (float)hr[t + 128] * w3[t + 128] +
            (float)hr[t + 256] * w3[t + 256];
  red[t] = s;
  __syncthreads();
  for (int k = 64; k > 0; k >>= 1) { if (t < k) red[t] += red[t + k]; __syncthreads(); }
  if (t == 0) {
    float l = red[0] + b3[0];
    probs[row] = 1.f / (1.f + expf(-l));
  }
}

// ---------------------------------------------------------------------------
// mask: binary = p>=0.5; if count<102 use stable top-102 (rank with index
// tie-break == jax.lax.top_k); mask[:,0]=True. Output 0.0/1.0 f32.
// ---------------------------------------------------------------------------
__global__ __launch_bounds__(1024) void mask_kernel(const float* __restrict__ probs,
                                                    float* __restrict__ maskf) {
  __shared__ float p[NTOK];
  __shared__ int scount;
  int b = blockIdx.x, i = threadIdx.x;
  if (i == 0) scount = 0;
  p[i] = probs[b * NTOK + i];
  __syncthreads();
  int binary = (p[i] >= 0.5f) ? 1 : 0;
  atomicAdd(&scount, binary);
  float pi = p[i];
  int rank = 0;
  for (int j = 0; j < NTOK; ++j) {
    float pj = p[j];
    rank += (pj > pi) || (pj == pi && j < i);
  }
  __syncthreads();
  int total = scount;
  bool m = (total < MIN_KEEP) ? (rank < MIN_KEEP) : (binary != 0);
  if (i == 0) m = true;
  maskf[b * NTOK + i] = m ? 1.f : 0.f;
}

// ---------------------------------------------------------------------------
// Masked attention per (b, h, 16-query tile). 4 waves / 128 threads.
// Phase 1: S = (q*0.125) @ k^T via WMMA, masked keys -> -inf, staged f16 in LDS
// Phase 2: softmax over 1024 keys (masked keys get exactly 0.0 weight, which
//          makes the reference's V-masking a numeric no-op)
// Phase 3: out = P @ V via WMMA; V staged per 128-key chunk in LDS through the
//          async global->LDS path, B-gather then hits LDS instead of HBM.
// ---------------------------------------------------------------------------
__global__ __launch_bounds__(128) void attn_kernel(const half_t* __restrict__ qkv,
                                                   const float* __restrict__ maskf,
                                                   half_t* __restrict__ attn_out) {
  __shared__ __align__(16) half_t S[16 * NTOK];   // 32 KB scores/probs
  __shared__ __align__(16) half_t VC[128 * HDIM]; // 16 KB V chunk
  __shared__ float red[16][8];
  __shared__ float rmax[16], rsum[16];

  int b = blockIdx.z, h = blockIdx.y, m0 = blockIdx.x * 16;
  int tid = threadIdx.x, wave = tid >> 5, lane = tid & 31;
  int lm = lane & 15, hs = lane >> 4;

  // q fragments (2 k-steps of 32 over HDIM=64), pre-scaled by HEAD_DIM^-0.5
  v16h aq[2];
  {
    const half_t* qr = qkv + (size_t)(b * NTOK + m0 + lm) * QKVC + h * HDIM + 8 * hs;
#pragma unroll
    for (int s = 0; s < 2; ++s) {
      v8h lo = *(const v8h*)(qr + 32 * s);
      v8h hi = *(const v8h*)(qr + 32 * s + 16);
#pragma unroll
      for (int i = 0; i < 8; ++i) {
        aq[s][i]     = (half_t)((float)lo[i] * 0.125f);
        aq[s][i + 8] = (half_t)((float)hi[i] * 0.125f);
      }
    }
  }

  // Phase 1: score tiles (64 key-tiles split across 4 waves)
  for (int jt = wave; jt < 64; jt += 4) {
    int key = jt * 16 + lm;
    const half_t* kr = qkv + (size_t)(b * NTOK + key) * QKVC + EMBED + h * HDIM + 8 * hs;
    v8f acc = {};
#pragma unroll
    for (int s = 0; s < 2; ++s) {
      v16h bk = frag_from_f16(kr + 32 * s);
      acc = __builtin_amdgcn_wmma_f32_16x16x32_f16(false, aq[s], false, bk, (short)0,
                                                   acc, false, false);
    }
    float mv = maskf[b * NTOK + key];
#pragma unroll
    for (int r = 0; r < 8; ++r) {
      float sv = (mv != 0.f) ? acc[r] : -__builtin_huge_valf();
      S[(r + 8 * hs) * NTOK + key] = (half_t)sv;
    }
  }
  __syncthreads();

  // Phase 2: softmax. thread -> (row = tid%16, 128-key chunk = tid/16)
  {
    int row = tid & 15, chunk = tid >> 4, base = chunk * 128;
    half_t* sr = S + row * NTOK;
    float mx = -__builtin_huge_valf();
    for (int k = 0; k < 128; ++k) mx = fmaxf(mx, (float)sr[base + k]);
    red[row][chunk] = mx;
    __syncthreads();
    if (chunk == 0) {
      float m = red[row][0];
#pragma unroll
      for (int c = 1; c < 8; ++c) m = fmaxf(m, red[row][c]);
      rmax[row] = m;
    }
    __syncthreads();
    float rm = rmax[row], s = 0.f;
    for (int k = 0; k < 128; ++k) {
      float e = expf((float)sr[base + k] - rm);
      s += e;
      sr[base + k] = (half_t)e;
    }
    red[row][chunk] = s;
    __syncthreads();
    if (chunk == 0) {
      float t2 = 0.f;
#pragma unroll
      for (int c = 0; c < 8; ++c) t2 += red[row][c];
      rsum[row] = t2;
    }
    __syncthreads();
    float inv = 1.f / rsum[row];
    for (int k = 0; k < 128; ++k) sr[base + k] = (half_t)((float)sr[base + k] * inv);
  }
  __syncthreads();

  // Phase 3: out(16x64) = P(16x1024) @ V(1024x64); wave -> 16-wide d tile.
  // V is staged chunk-by-chunk (128 keys x 64 dims) into LDS.
  {
    int d0 = wave * 16;
    v8f acc = {};
    for (int c = 0; c < 8; ++c) {
      // stage chunk: 128 rows x 64 halfs = 1024 x 16B transfers / 128 threads
      for (int i = tid; i < 1024; i += 128) {
        int row = i >> 3, vv = i & 7;
        copy16_to_lds(qkv + (size_t)(b * NTOK + c * 128 + row) * QKVC + 2 * EMBED +
                          h * HDIM + vv * 8,
                      VC + row * HDIM + vv * 8);
      }
      wait_lds_copies();
#pragma unroll
      for (int kk = 0; kk < 128; kk += 32) {
        v16h aP = frag_from_f16(S + lm * NTOK + c * 128 + kk + 8 * hs);
        v16h bV;
#pragma unroll
        for (int e = 0; e < 8; ++e) {
          bV[e]     = VC[(kk + 8 * hs + e) * HDIM + d0 + lm];
          bV[e + 8] = VC[(kk + 16 + 8 * hs + e) * HDIM + d0 + lm];
        }
        acc = __builtin_amdgcn_wmma_f32_16x16x32_f16(false, aP, false, bV, (short)0,
                                                     acc, false, false);
      }
      __syncthreads();   // protect VC before restaging
    }
#pragma unroll
    for (int r = 0; r < 8; ++r) {
      size_t o = (size_t)(b * NTOK + m0 + r + 8 * hs) * EMBED + h * HDIM + d0 + lm;
      attn_out[o] = (half_t)acc[r];
    }
  }
}

// ---------------------------------------------------------------------------
// host-side orchestration
// ---------------------------------------------------------------------------
extern "C" void kernel_launch(void* const* d_in, const int* in_sizes, int n_in,
                              void* d_out, int out_size, void* d_ws, size_t ws_size,
                              hipStream_t stream) {
  (void)in_sizes; (void)n_in; (void)out_size; (void)ws_size;
  const float* tokens   = (const float*)d_in[0];
  const int*   lidx     = (const int*)d_in[1];
  const float* layeremb = (const float*)d_in[2];
  const float* ln_g     = (const float*)d_in[3];
  const float* ln_b     = (const float*)d_in[4];
  const float* w1       = (const float*)d_in[5];
  const float* b1       = (const float*)d_in[6];
  const float* w2       = (const float*)d_in[7];
  const float* b2       = (const float*)d_in[8];
  const float* w3       = (const float*)d_in[9];
  const float* b3       = (const float*)d_in[10];
  const float* qkv_w    = (const float*)d_in[11];
  const float* qkv_b    = (const float*)d_in[12];
  const float* proj_w   = (const float*)d_in[13];
  const float* proj_b   = (const float*)d_in[14];
  float* out = (float*)d_out;

  char* w = (char*)d_ws;
  float*  tmean  = (float*) (w + 0);              //     24576 B
  half_t* tok_h  = (half_t*)(w + 24576);          //  12582912 B
  half_t* xln    = (half_t*)(w + 12607488);       //  12582912 B
  half_t* h1     = (half_t*)(w + 25190400);       //   6291456 B
  half_t* h2     = (half_t*)(w + 31481856);       //   6291456 B
  float*  probs  = (float*) (w + 37773312);       //     32768 B
  float*  maskf  = (float*) (w + 37806080);       //     32768 B
  half_t* qkvh   = (half_t*)(w + 37838848);       //  37748736 B
  half_t* attnh  = (half_t*)(w + 75587584);       //  12582912 B
  half_t* w1h    = (half_t*)(w + 88170496);       //    589824 B
  half_t* w2h    = (half_t*)(w + 88760320);       //    294912 B
  half_t* qkvwh  = (half_t*)(w + 89055232);       //   3538944 B
  half_t* projwh = (half_t*)(w + 92594176);       //   1179648 B (end 93773824)

  // f16 weight copies (consumed by the LDS-staged WMMA GEMMs)
  cast_kernel<<<(HIDDEN * EMBED + 255) / 256, 256, 0, stream>>>(w1, w1h,
                                                                HIDDEN * EMBED);
  cast_kernel<<<(HIDDEN * HIDDEN + 255) / 256, 256, 0, stream>>>(w2, w2h,
                                                                 HIDDEN * HIDDEN);
  cast_kernel<<<(QKVC * EMBED + 255) / 256, 256, 0, stream>>>(qkv_w, qkvwh,
                                                              QKVC * EMBED);
  cast_kernel<<<(EMBED * EMBED + 255) / 256, 256, 0, stream>>>(proj_w, projwh,
                                                               EMBED * EMBED);
  cast_kernel<<<(ROWS * EMBED + 255) / 256, 256, 0, stream>>>(tokens, tok_h,
                                                              ROWS * EMBED);

  // actor path
  rowmean_kernel<<<BATCH, 256, 0, stream>>>(tokens, tmean);
  ln_kernel<<<ROWS, 256, 0, stream>>>(tokens, tmean, layeremb, lidx, ln_g, ln_b, xln);
  gemm_kernel<true, true, false, EMBED>
      <<<dim3(HIDDEN / 16, ROWS / 128), 256, 0, stream>>>(xln, w1h, b1, h1, nullptr,
                                                          nullptr, ROWS, HIDDEN);
  gemm_kernel<true, true, false, HIDDEN>
      <<<dim3(HIDDEN / 16, ROWS / 128), 256, 0, stream>>>(h1, w2h, b2, h2, nullptr,
                                                          nullptr, ROWS, HIDDEN);
  logits_kernel<<<ROWS, 128, 0, stream>>>(h2, w3, b3, probs);
  mask_kernel<<<BATCH, 1024, 0, stream>>>(probs, maskf);

  // attention path
  gemm_kernel<false, true, false, EMBED>
      <<<dim3(QKVC / 16, ROWS / 128), 256, 0, stream>>>(tok_h, qkvwh, qkv_b, qkvh,
                                                        nullptr, nullptr, ROWS, QKVC);
  attn_kernel<<<dim3(NTOK / 16, NHEAD, BATCH), 128, 0, stream>>>(qkvh, maskf, attnh);
  gemm_kernel<false, false, true, EMBED>
      <<<dim3(EMBED / 16, ROWS / 128), 256, 0, stream>>>(attnh, projwh, proj_b,
                                                         nullptr, out, maskf, ROWS,
                                                         EMBED);
}
